// DendriticLayer_39737037422942
// MI455X (gfx1250) — compile-verified, matrix operand order checked by reading, not run on hardware
//
#include <hip/hip_runtime.h>

// ---------- CDNA5 wave32 / WMMA / TDM types ----------
typedef __bf16 bf16;
typedef __attribute__((ext_vector_type(16))) __bf16 v16bf;
typedef __attribute__((ext_vector_type(8)))  float  v8f;
typedef unsigned int u32x4 __attribute__((ext_vector_type(4)));
typedef int          i32x8 __attribute__((ext_vector_type(8)));
typedef int          i32x4 __attribute__((ext_vector_type(4)));

#define IN_DIM     1024
#define BATCH      4096
#define N_NEURONS  1024
#define N_DEND     16
#define N_SOMA     16384
#define SLOPE      0.1f

#define KTILES     32                    // k-tiles of K=32
#define BTILES     (BATCH / 16)          // 256 batch tiles
#define STILES     (N_SOMA / 16)         // 1024 soma tiles (1 neuron each)
#define NG         4                     // neurons per block
#define CHUNK      4                     // k-tiles per LDS stage
#define NSTAGE     (KTILES / CHUNK)      // 8 stages
#define FRAG_ELEMS 512                   // bf16 per 16x32 fragment (32 lanes x 16)
#define STAGE_ELEMS (NG * CHUNK * FRAG_ELEMS)   // 8192 bf16 = 16 KB per stage

#if __has_builtin(__builtin_amdgcn_tensor_load_to_lds) && __has_builtin(__builtin_amdgcn_s_wait_tensorcnt)
#define USE_TDM 1
#else
#define USE_TDM 0
#endif

__device__ __forceinline__ float leaky(float v) { return v >= 0.f ? v : SLOPE * v; }

// ---------------------------------------------------------------------------
// Prep 1: x (f32 row-major) -> bf16 WMMA A-fragments (lane-contiguous order).
// ---------------------------------------------------------------------------
__global__ __launch_bounds__(128) void pack_x_kernel(const float* __restrict__ x,
                                                     bf16* __restrict__ xw) {
    const int frag = blockIdx.x * 4 + (threadIdx.x >> 5);
    const int lane = threadIdx.x & 31;
    const int ib   = frag >> 5;
    const int kk   = frag & 31;
    const int row  = ib * 16 + (lane & 15);
    const int kb   = kk * 32 + ((lane >> 4) << 3);
    const float* src = x + (size_t)row * IN_DIM + kb;
    v16bf v;
#pragma unroll
    for (int t = 0; t < 8; ++t) {
        v[t]     = (bf16)src[t];
        v[t + 8] = (bf16)src[16 + t];
    }
    *(v16bf*)(xw + (size_t)frag * FRAG_ELEMS + lane * 16) = v;
}

// ---------------------------------------------------------------------------
// Prep 2: (Wd * dendrite_mask) -> bf16 WMMA B-fragments, mask baked in.
// ---------------------------------------------------------------------------
__global__ __launch_bounds__(128) void pack_w_kernel(const float* __restrict__ Wd,
                                                     const float* __restrict__ dmask,
                                                     bf16* __restrict__ bw) {
    const int frag = blockIdx.x * 4 + (threadIdx.x >> 5);
    const int lane = threadIdx.x & 31;
    const int st   = frag >> 5;
    const int kk   = frag & 31;
    const int nrow = st * 16 + (lane & 15);
    const int kb   = kk * 32 + ((lane >> 4) << 3);
    const size_t off = (size_t)nrow * IN_DIM + kb;
    v16bf v;
#pragma unroll
    for (int t = 0; t < 8; ++t) {
        v[t]     = (bf16)(Wd[off + t]      * dmask[off + t]);
        v[t + 8] = (bf16)(Wd[off + 16 + t] * dmask[off + 16 + t]);
    }
    *(v16bf*)(bw + (size_t)frag * FRAG_ELEMS + lane * 16) = v;
}

// ---------------------------------------------------------------------------
// Prep 3: block-diagonal extract of Ws (soma_mask is block-diagonal).
// ---------------------------------------------------------------------------
__global__ __launch_bounds__(256) void pack_wsb_kernel(const float* __restrict__ Ws,
                                                       const float* __restrict__ smask,
                                                       float* __restrict__ wsb) {
    const int i = blockIdx.x * blockDim.x + threadIdx.x;
    if (i < N_NEURONS * N_DEND) {
        const int n = i >> 4;
        const size_t col = (size_t)n * N_SOMA + i;
        wsb[i] = Ws[col] * smask[col];
    }
}

// ---------------------------------------------------------------------------
// Fused main kernel.
// Block = 4 waves on the SAME 4-neuron group, each wave a different batch
// tile. B fragments staged per block in LDS via TDM (double buffered, copy of
// stage s+1 overlaps compute of stage s). A fragments clause-loaded per wave.
// B fragments software-pipelined one k-step ahead in distinct registers so
// ds_load latency overlaps WMMA issue. Layer 2 (block-diagonal Ws) is an
// in-register wave32 shuffle reduction on the WMMA accumulator.
// ---------------------------------------------------------------------------
__global__ __launch_bounds__(128) void fused_dendrite_kernel(
        const bf16*  __restrict__ xw,
        const bf16*  __restrict__ bw,
        const float* __restrict__ bd,
        const float* __restrict__ wsb,
        const float* __restrict__ bs,
        float*       __restrict__ out) {
    __shared__ bf16 lsB[2][STAGE_ELEMS];     // 2 x 16 KB

    const int lane = threadIdx.x & 31;
    const int wave = threadIdx.x >> 5;
    const int ng   = blockIdx.x & (N_NEURONS / NG - 1);   // neuron group 0..255
    const int ibg  = blockIdx.x >> 8;                     // 0..63
    const int ib   = ibg * 4 + wave;                      // this wave's batch tile

    const bf16* ap = xw + (size_t)ib * (KTILES * FRAG_ELEMS) + lane * 16;
    const size_t bframe = (size_t)(ng * NG) * (KTILES * FRAG_ELEMS);

#if USE_TDM
    // One TDM descriptor copies the whole 16 KB stage: 2D tile of NG rows
    // (neurons) x CHUNK*512 contiguous bf16, global row stride = KTILES*512.
    auto tdm_issue = [&](int buf, int t0) {
        unsigned long long ga =
            (unsigned long long)(size_t)(bw + bframe + (size_t)t0 * FRAG_ELEMS);
        unsigned lds = (unsigned)(size_t)(&lsB[buf][0]);
        u32x4 g0;
        g0[0] = 1u;                                        // count=1, user mode
        g0[1] = lds;                                       // lds_addr (bytes)
        g0[2] = (unsigned)(ga & 0xFFFFFFFFu);              // global_addr[31:0]
        g0[3] = (unsigned)((ga >> 32) & 0x1FFFFFFu) | (2u << 30); // addr hi | type=2
        i32x8 g1;
        g1[0] = (1 << 16);                                 // data_size = 2 bytes
        g1[1] = (int)((unsigned)(CHUNK * FRAG_ELEMS) << 16);      // tensor_dim0 lo16
        g1[2] = (int)(((CHUNK * FRAG_ELEMS) >> 16) | (NG << 16)); // dim0 hi | dim1 lo
        g1[3] = (int)((unsigned)(CHUNK * FRAG_ELEMS) << 16);      // dim1 hi(0) | tile_dim0
        g1[4] = NG;                                        // tile_dim1=NG, tile_dim2=0
        g1[5] = KTILES * FRAG_ELEMS;                       // tensor_dim0_stride = 16384
        g1[6] = 0;
        g1[7] = 0;
        i32x4 z4 = {0, 0, 0, 0};                           // D# groups 2/3 unused (2D)
        i32x8 z8 = {0, 0, 0, 0, 0, 0, 0, 0};
        __builtin_amdgcn_tensor_load_to_lds(g0, g1, z4, z4, z8, 0);
    };
    if (wave == 0) tdm_issue(0, 0);
#endif

    v8f c0 = {}, c1 = {}, c2 = {}, c3 = {};

    for (int s = 0; s < NSTAGE; ++s) {
        const int buf = s & 1;
        const int t0  = s * CHUNK;
#if USE_TDM
        if (wave == 0) __builtin_amdgcn_s_wait_tensorcnt(0);
        __syncthreads();                     // stage s data visible to all waves
        if (wave == 0 && s + 1 < NSTAGE) tdm_issue(buf ^ 1, t0 + CHUNK);
#else
        __syncthreads();                     // everyone done reading this buffer
        {   // cooperative fallback copy: 1024 uint4 per stage, 8 per thread
            const uint4* gsrc = (const uint4*)(bw + bframe);
            uint4* dst = (uint4*)&lsB[buf][0];
            for (int u = threadIdx.x; u < STAGE_ELEMS / 8; u += 128) {
                const int j = u >> 8;        // neuron row (256 uint4 each)
                const int r = u & 255;
                dst[u] = gsrc[(size_t)j * (KTILES * FRAG_ELEMS / 8)
                              + (size_t)t0 * (FRAG_ELEMS / 8) + r];
            }
        }
        __syncthreads();
#endif
        // Prefetch this chunk's 4 A fragments (clause of global_load_b128,
        // L2-resident) so their latency is amortized over 16 WMMAs.
        v16bf a0 = *(const v16bf*)(ap + (size_t)(t0 + 0) * FRAG_ELEMS);
        v16bf a1 = *(const v16bf*)(ap + (size_t)(t0 + 1) * FRAG_ELEMS);
        v16bf a2 = *(const v16bf*)(ap + (size_t)(t0 + 2) * FRAG_ELEMS);
        v16bf a3 = *(const v16bf*)(ap + (size_t)(t0 + 3) * FRAG_ELEMS);
        const bf16* lb = &lsB[buf][lane * 16];

        // Software pipeline B one k-step ahead in distinct registers so the
        // next 8 ds_load_b128 are in flight while this step's WMMAs issue.
        v16bf bb0 = *(const v16bf*)(lb + (0 * CHUNK + 0) * FRAG_ELEMS);
        v16bf bb1 = *(const v16bf*)(lb + (1 * CHUNK + 0) * FRAG_ELEMS);
        v16bf bb2 = *(const v16bf*)(lb + (2 * CHUNK + 0) * FRAG_ELEMS);
        v16bf bb3 = *(const v16bf*)(lb + (3 * CHUNK + 0) * FRAG_ELEMS);
#pragma unroll
        for (int c = 0; c < CHUNK; ++c) {
            v16bf nb0, nb1, nb2, nb3;
            if (c + 1 < CHUNK) {
                nb0 = *(const v16bf*)(lb + (0 * CHUNK + c + 1) * FRAG_ELEMS);
                nb1 = *(const v16bf*)(lb + (1 * CHUNK + c + 1) * FRAG_ELEMS);
                nb2 = *(const v16bf*)(lb + (2 * CHUNK + c + 1) * FRAG_ELEMS);
                nb3 = *(const v16bf*)(lb + (3 * CHUNK + c + 1) * FRAG_ELEMS);
            }
            v16bf a = (c == 0) ? a0 : (c == 1) ? a1 : (c == 2) ? a2 : a3;
            c0 = __builtin_amdgcn_wmma_f32_16x16x32_bf16(false, a, false, bb0, (short)0, c0, false, false);
            c1 = __builtin_amdgcn_wmma_f32_16x16x32_bf16(false, a, false, bb1, (short)0, c1, false, false);
            c2 = __builtin_amdgcn_wmma_f32_16x16x32_bf16(false, a, false, bb2, (short)0, c2, false, false);
            c3 = __builtin_amdgcn_wmma_f32_16x16x32_bf16(false, a, false, bb3, (short)0, c3, false, false);
            if (c + 1 < CHUNK) {
                bb0 = nb0; bb1 = nb1; bb2 = nb2; bb3 = nb3;
            }
        }
    }

    // ---- fused layer 2 epilogue --------------------------------------------
    // C layout: N = lane&15 (dendrite d), M = vgpr + 8*(lane>>4).
    const int col   = lane & 15;
    const int mbase = (lane >> 4) << 3;
    v8f cs[NG] = {c0, c1, c2, c3};

#pragma unroll
    for (int j = 0; j < NG; ++j) {
        const int   n   = ng * NG + j;
        const float bdv = bd[n * N_DEND + col];
        const float wv  = wsb[n * N_DEND + col];
        float acc[8];
#pragma unroll
        for (int r = 0; r < 8; ++r)
            acc[r] = leaky(cs[j][r] + bdv) * wv;
        // xor masks 1,2,4,8 stay inside each 16-lane half of the wave32
#pragma unroll
        for (int off = 1; off < 16; off <<= 1)
#pragma unroll
            for (int r = 0; r < 8; ++r)
                acc[r] += __shfl_xor(acc[r], off, 32);
        if (col == 0) {
            const float bsv = bs[n];
            float* o = out + (size_t)(ib * 16 + mbase) * N_NEURONS + n;
#pragma unroll
            for (int r = 0; r < 8; ++r)
                o[(size_t)r * N_NEURONS] = leaky(acc[r] + bsv);
        }
    }
}

// ---------------------------------------------------------------------------
extern "C" void kernel_launch(void* const* d_in, const int* in_sizes, int n_in,
                              void* d_out, int out_size, void* d_ws, size_t ws_size,
                              hipStream_t stream) {
    const float* x     = (const float*)d_in[0];
    const float* Wd    = (const float*)d_in[1];
    const float* bd    = (const float*)d_in[2];
    const float* Ws    = (const float*)d_in[3];
    const float* bs    = (const float*)d_in[4];
    const float* dmask = (const float*)d_in[5];
    const float* smask = (const float*)d_in[6];
    float* out = (float*)d_out;

    // workspace: xw (8 MB bf16) | bw (32 MB bf16) | wsb (64 KB f32)
    bf16*  xw  = (bf16*)d_ws;
    bf16*  bw  = xw + (size_t)BATCH * IN_DIM;
    float* wsb = (float*)(bw + (size_t)N_SOMA * IN_DIM);

    hipLaunchKernelGGL(pack_x_kernel,   dim3(BTILES * KTILES / 4), dim3(128), 0, stream, x, xw);
    hipLaunchKernelGGL(pack_w_kernel,   dim3(STILES * KTILES / 4), dim3(128), 0, stream, Wd, dmask, bw);
    hipLaunchKernelGGL(pack_wsb_kernel, dim3(64), dim3(256), 0, stream, Ws, smask, wsb);

    // 256 neuron groups x 64 batch-tile groups; block = 4 waves
    hipLaunchKernelGGL(fused_dendrite_kernel, dim3((N_NEURONS / NG) * (BTILES / 4)),
                       dim3(128), 0, stream, xw, bw, bd, wsb, bs, out);
}
